// NonLinearSelfAttention_59794534695069
// MI455X (gfx1250) — matrix operand order repaired and checked by reading, not compiled, not run on hardware
//
#include <hip/hip_runtime.h>
#include <hip/hip_bf16.h>

typedef __attribute__((ext_vector_type(16))) __bf16 v16bf;
typedef __attribute__((ext_vector_type(8)))  __bf16 v8bf;
typedef __attribute__((ext_vector_type(8)))  float  v8f;
typedef __attribute__((ext_vector_type(4)))  float  f4;

#define DIMC   128
#define SEQ    4096
#define BATCH  8
#define BM     128      // q rows per block
#define BN     64       // kv rows per tile
#define NT     (SEQ / BN)
#define XRM_S  136      // row-major x tile stride (bf16 elems), padded
#define XT_S   72       // transposed x tile stride (bf16 elems), padded
#define ON_S   136      // O_norm tile stride (bf16 elems), padded
// one buffer = BN*XRM_S + DIMC*XT_S = 8704 + 9216 = 17920 bf16 (35840 B)
#define BUF_ELEMS 17920
// double-buffered for async pipelining; proj-phase On (17408 bf16) aliases buf0
#define SMEM_ELEMS (2 * BUF_ELEMS)

// ---------------------------------------------------------------------------
// Pre-pass: x fp32 -> bf16 in two layouts: xbf[B][N][D] and xT[B][D][N].
// Transpose staged through LDS so both global writes are coalesced.
// ---------------------------------------------------------------------------
__global__ __launch_bounds__(256) void convert_kernel(
    const float* __restrict__ x, __bf16* __restrict__ xbf, __bf16* __restrict__ xT)
{
    __shared__ __align__(16) __bf16 tt[DIMC * XT_S];
    const int tid = threadIdx.x;
    const int b   = blockIdx.x / (SEQ / BN);
    const int nt  = blockIdx.x % (SEQ / BN);
    const int n0  = nt * BN;

    const int kv    = tid & 63;
    const int dbase = (tid >> 6) * 32;
    const float* src = x + ((size_t)b * SEQ + n0 + kv) * DIMC + dbase;
    __bf16 bv[32];
    #pragma unroll
    for (int j = 0; j < 8; ++j) {
        f4 v = *(const f4*)(src + 4 * j);
        #pragma unroll
        for (int i = 0; i < 4; ++i) bv[4 * j + i] = (__bf16)v[i];
    }
    __bf16* dst = xbf + ((size_t)b * SEQ + n0 + kv) * DIMC + dbase;
    #pragma unroll
    for (int j = 0; j < 4; ++j)
        *(v8bf*)(dst + 8 * j) = *(const v8bf*)(bv + 8 * j);
    #pragma unroll
    for (int i = 0; i < 32; ++i)
        tt[(dbase + i) * XT_S + kv] = bv[i];
    __syncthreads();
    const int d = tid >> 1, half = tid & 1;
    __bf16* dstT = xT + ((size_t)b * DIMC + d) * SEQ + n0 + half * 32;
    const __bf16* srcT = tt + d * XT_S + half * 32;
    #pragma unroll
    for (int j = 0; j < 4; ++j)
        *(v8bf*)(dstT + 8 * j) = *(const v8bf*)(srcT + 8 * j);
}

// ---------------------------------------------------------------------------
// Fused flash-attention + projection.  ASYNC=true: double-buffered LDS tiles
// streamed with global_load_async_to_lds_b128 underneath the WMMA pipeline.
// ---------------------------------------------------------------------------
template <bool ASYNC>
__global__ __launch_bounds__(256) void fattn_proj_kernel(
    const float* __restrict__ x, const float* __restrict__ W,
    const float* __restrict__ bias, float* __restrict__ out,
    const __bf16* __restrict__ xbf, const __bf16* __restrict__ xT)
{
    __shared__ __align__(16) __bf16 smem[SMEM_ELEMS];
    __bf16* on = smem;                   // [BM][ON_S] normalized O (aliases buf0)

    const int tid  = threadIdx.x;
    const int lane = tid & 31;
    const int wid  = tid >> 5;           // 8 waves
    const int lhi  = lane >> 4;          // which 16-lane half
    const int l16  = lane & 15;

    const int batch = blockIdx.x / (SEQ / BM);
    const int qtile = blockIdx.x % (SEQ / BM);
    const int n0q   = qtile * BM;
    const float*  xb  = x   + (size_t)batch * SEQ * DIMC;
    const __bf16* xbb = xbf + (size_t)batch * SEQ * DIMC;

    const float csc = 0.08838834764831845f * 1.4426950408889634f; // 1/sqrt(128)*log2(e)

    // ---- Q B-fragments (B = Q^T, K=dim): lane = q col, 16 consecutive dims ----
    v16bf qf[4];
    if constexpr (ASYNC) {
        const __bf16* qrow = xbb + (size_t)(n0q + wid * 16 + l16) * DIMC;
        #pragma unroll
        for (int k = 0; k < 4; ++k)
            qf[k] = *(const v16bf*)(qrow + k * 32 + lhi * 16);
    } else {
        const float* qrow = xb + (size_t)(n0q + wid * 16 + l16) * DIMC;
        #pragma unroll
        for (int k = 0; k < 4; ++k) {
            const float* src = qrow + k * 32 + lhi * 16;
            #pragma unroll
            for (int j = 0; j < 4; ++j) {
                f4 v = *(const f4*)(src + 4 * j);
                #pragma unroll
                for (int i = 0; i < 4; ++i) qf[k][4 * j + i] = (__bf16)v[i];
            }
        }
    }

    // async-loader per-thread constants: 4 b128 segments per tile per thread
    uint32_t ldsRM[4], ldsXT[4], voRM[4], voXT[4];
    uint64_t saRM = 0, saXT = 0;
    if constexpr (ASYNC) {
        #pragma unroll
        for (int j = 0; j < 4; ++j) {
            const int seg = tid + 256 * j;            // 0..1023
            const int row = seg >> 4, c16 = seg & 15; // RM: 64 rows x 16 segs
            ldsRM[j] = (uint32_t)(uintptr_t)(smem + row * XRM_S + c16 * 8);
            voRM[j]  = (uint32_t)((row * DIMC + c16 * 8) * 2);
            const int d = seg >> 3, s8 = seg & 7;     // XT: 128 rows x 8 segs
            ldsXT[j] = (uint32_t)(uintptr_t)(smem + BN * XRM_S + d * XT_S + s8 * 8);
            voXT[j]  = (uint32_t)(((size_t)d * SEQ + s8 * 8) * 2);
        }
        saRM = (uint64_t)(uintptr_t)xbb;
        saXT = (uint64_t)(uintptr_t)(xT + (size_t)batch * DIMC * SEQ);
    }

    v8f oacc[8];
    #pragma unroll
    for (int d = 0; d < 8; ++d) { v8f z = {}; oacc[d] = z; }
    float m = -__builtin_inff();
    float l = 0.0f;

    // prologue: stream tile 0 into buffer 0
    if constexpr (ASYNC) {
        #pragma unroll
        for (int j = 0; j < 4; ++j)
            asm volatile("global_load_async_to_lds_b128 %0, %1, %2"
                         :: "v"(ldsRM[j]), "v"(voRM[j]), "s"(saRM) : "memory");
        #pragma unroll
        for (int j = 0; j < 4; ++j)
            asm volatile("global_load_async_to_lds_b128 %0, %1, %2"
                         :: "v"(ldsXT[j]), "v"(voXT[j]), "s"(saXT) : "memory");
    }

    for (int it = 0; it < NT; ++it) {
        const __bf16* xrm_c;
        const __bf16* xt_c;
        if constexpr (ASYNC) {
            const int cur = it & 1;
            xrm_c = smem + cur * BUF_ELEMS;
            xt_c  = xrm_c + BN * XRM_S;
            asm volatile("s_wait_asynccnt 0x0" ::: "memory"); // tile it resident
            __syncthreads();                                  // all shares visible
            if (it + 1 < NT) {                                // stream tile it+1
                const uint32_t bufoff = (uint32_t)(((it + 1) & 1) * BUF_ELEMS * 2);
                const uint32_t baseRM = (uint32_t)((it + 1) * BN * DIMC * 2);
                const uint32_t baseXT = (uint32_t)((it + 1) * BN * 2);
                #pragma unroll
                for (int j = 0; j < 4; ++j) {
                    uint32_t la = ldsRM[j] + bufoff, vr = voRM[j] + baseRM;
                    asm volatile("global_load_async_to_lds_b128 %0, %1, %2"
                                 :: "v"(la), "v"(vr), "s"(saRM) : "memory");
                }
                #pragma unroll
                for (int j = 0; j < 4; ++j) {
                    uint32_t la = ldsXT[j] + bufoff, vx = voXT[j] + baseXT;
                    asm volatile("global_load_async_to_lds_b128 %0, %1, %2"
                                 :: "v"(la), "v"(vx), "s"(saXT) : "memory");
                }
            }
        } else {
            xrm_c = smem;
            xt_c  = smem + BN * XRM_S;
            const int n0k = it * BN;
            __syncthreads();
            {   // fallback: fp32 load + convert + dual-layout LDS store
                const int kv    = tid & 63;
                const int dbase = (tid >> 6) * 32;
                const float* src = xb + (size_t)(n0k + kv) * DIMC + dbase;
                __bf16 bv[32];
                #pragma unroll
                for (int j = 0; j < 8; ++j) {
                    f4 v = *(const f4*)(src + 4 * j);
                    #pragma unroll
                    for (int i = 0; i < 4; ++i) bv[4 * j + i] = (__bf16)v[i];
                }
                #pragma unroll
                for (int j = 0; j < 4; ++j)
                    *(v8bf*)((__bf16*)xrm_c + kv * XRM_S + dbase + 8 * j) =
                        *(const v8bf*)(bv + 8 * j);
                #pragma unroll
                for (int i = 0; i < 32; ++i)
                    ((__bf16*)xt_c)[(dbase + i) * XT_S + kv] = bv[i];
                if (it + 1 < NT)
                    __builtin_prefetch(src + BN * DIMC, 0, 1);
            }
            __syncthreads();
        }

        // ---- St = X_tile * Q^T  (St[kv][q]; C-layout == A-layout for P) ----
        v8f st[4];
        #pragma unroll
        for (int t = 0; t < 4; ++t) {
            v8f c = {};
            #pragma unroll
            for (int k = 0; k < 4; ++k) {
                const __bf16* arow = xrm_c + (t * 16 + l16) * XRM_S + k * 32 + lhi * 16;
                v8bf alo = *(const v8bf*)(arow);
                v8bf ahi = *(const v8bf*)(arow + 8);
                v16bf a = __builtin_shufflevector(alo, ahi,
                          0,1,2,3,4,5,6,7,8,9,10,11,12,13,14,15);
                c = __builtin_amdgcn_wmma_f32_16x16x32_bf16(
                        false, a, false, qf[k], (short)0, c, false, false);
            }
            st[t] = c;
        }

        // ---- online softmax (per q = lane%16; halves combined via shfl_xor 16) ----
        float tmax = -__builtin_inff();
        #pragma unroll
        for (int t = 0; t < 4; ++t)
            #pragma unroll
            for (int r = 0; r < 8; ++r) tmax = fmaxf(tmax, st[t][r]);
        tmax = fmaxf(tmax, __shfl_xor(tmax, 16, 32));
        const float mnew  = fmaxf(m, tmax);
        const float alpha = __builtin_amdgcn_exp2f((m - mnew) * csc);
        float rowsum = 0.0f;
        v16bf pa[2];
        #pragma unroll
        for (int t = 0; t < 4; ++t)
            #pragma unroll
            for (int r = 0; r < 8; ++r) {
                float p = __builtin_amdgcn_exp2f((st[t][r] - mnew) * csc);
                rowsum += p;
                pa[t >> 1][((t & 1) << 3) + r] = (__bf16)p;  // P already in A-layout
            }
        rowsum += __shfl_xor(rowsum, 16, 32);
        l = l * alpha + rowsum;
        m = mnew;

        // per-accumulator-row rescale factors (q = r + 8*half)
        float af[8];
        #pragma unroll
        for (int r = 0; r < 8; ++r) af[r] = __shfl(alpha, r + (lhi << 3), 32);

        // ---- O += P * V  (B fragments from transposed LDS tile) ----
        #pragma unroll
        for (int d = 0; d < 8; ++d) {
            v8f c = oacc[d];
            #pragma unroll
            for (int r = 0; r < 8; ++r) c[r] *= af[r];
            #pragma unroll
            for (int k = 0; k < 2; ++k) {
                const __bf16* bp = xt_c + (d * 16 + l16) * XT_S + k * 32 + lhi * 8;
                v8bf blo = *(const v8bf*)(bp);
                v8bf bhi = *(const v8bf*)(bp + 16);
                v16bf bfr = __builtin_shufflevector(blo, bhi,
                          0,1,2,3,4,5,6,7,8,9,10,11,12,13,14,15);
                c = __builtin_amdgcn_wmma_f32_16x16x32_bf16(
                        false, pa[k], false, bfr, (short)0, c, false, false);
            }
            oacc[d] = c;
        }
    }

    // ---- normalize O and stage to LDS (reuses buffer 0) ----
    __syncthreads();
    {
        const float linv = 1.0f / l;
        float lf[8];
        #pragma unroll
        for (int r = 0; r < 8; ++r) lf[r] = __shfl(linv, r + (lhi << 3), 32);
        #pragma unroll
        for (int d = 0; d < 8; ++d)
            #pragma unroll
            for (int r = 0; r < 8; ++r)
                on[(wid * 16 + r + lhi * 8) * ON_S + d * 16 + l16] =
                    (__bf16)(oacc[d][r] * lf[r]);
    }
    __syncthreads();

    // ---- projection: F = O_norm * W^T + b (W fragments stream from L2) ----
    float* outp = out + (size_t)(batch * SEQ + n0q) * DIMC;
    #pragma unroll
    for (int ot = 0; ot < 8; ++ot) {
        const float bb = bias[ot * 16 + l16];
        v8f c;
        #pragma unroll
        for (int r = 0; r < 8; ++r) c[r] = bb;
        #pragma unroll
        for (int k = 0; k < 4; ++k) {
            const __bf16* ap = on + (wid * 16 + l16) * ON_S + k * 32 + lhi * 16;
            v8bf alo = *(const v8bf*)(ap);
            v8bf ahi = *(const v8bf*)(ap + 8);
            v16bf a = __builtin_shufflevector(alo, ahi,
                      0,1,2,3,4,5,6,7,8,9,10,11,12,13,14,15);
            const float* wrow = W + (size_t)(ot * 16 + l16) * DIMC + k * 32 + lhi * 16;
            v16bf wb;
            #pragma unroll
            for (int j = 0; j < 4; ++j) {
                f4 v = *(const f4*)(wrow + 4 * j);
                #pragma unroll
                for (int i = 0; i < 4; ++i) wb[4 * j + i] = (__bf16)v[i];
            }
            c = __builtin_amdgcn_wmma_f32_16x16x32_bf16(
                    false, a, false, wb, (short)0, c, false, false);
        }
        #pragma unroll
        for (int r = 0; r < 8; ++r)
            outp[(size_t)(wid * 16 + r + lhi * 8) * DIMC + ot * 16 + l16] = c[r];
    }
}

extern "C" void kernel_launch(void* const* d_in, const int* in_sizes, int n_in,
                              void* d_out, int out_size, void* d_ws, size_t ws_size,
                              hipStream_t stream) {
    const float* x = (const float*)d_in[0];
    const float* W = (const float*)d_in[1];
    const float* b = (const float*)d_in[2];
    float* out     = (float*)d_out;
    (void)in_sizes; (void)n_in; (void)out_size;

    dim3 grid(BATCH * (SEQ / BM));
    dim3 block(256);
    const size_t need = (size_t)2 * BATCH * SEQ * DIMC * sizeof(__hip_bfloat16);
    if (d_ws != nullptr && ws_size >= need) {
        __bf16* xbf = (__bf16*)d_ws;
        __bf16* xT  = xbf + (size_t)BATCH * SEQ * DIMC;
        hipLaunchKernelGGL(convert_kernel, dim3(BATCH * (SEQ / BN)), block, 0, stream,
                           x, xbf, xT);
        hipLaunchKernelGGL((fattn_proj_kernel<true>), grid, block, 0, stream,
                           x, W, b, out, xbf, xT);
    } else {
        hipLaunchKernelGGL((fattn_proj_kernel<false>), grid, block, 0, stream,
                           x, W, b, out, nullptr, nullptr);
    }
}
